// VoxelRCNNHead_89069031784635
// MI455X (gfx1250) — compile-verified, matrix operand order checked by reading, not run on hardware
//
#include <hip/hip_runtime.h>

typedef __attribute__((ext_vector_type(16))) __bf16          v16bf;
typedef __attribute__((ext_vector_type(8)))  float           v8f;

#define NZv 5
#define NYv 100
#define NXv 176
#define NVOX_GRID (NZv * NYv * NXv)   // 88000
#define N_ROIS 256
#define NPT 216                        // 6^3
#define KPRE 6912                      // 216*32
#define NTASK (N_ROIS * NPT)           // 55296

union BFrag  { uint4 q[2]; v16bf v; };          // 16 bf16 = two 16B blocks
union BPack  { __bf16 h[2]; unsigned int u; };  // packed bf16 pair

__device__ __forceinline__ v8f wmma_bf16(v16bf a, v16bf b, v8f c) {
    return __builtin_amdgcn_wmma_f32_16x16x32_bf16(false, a, false, b,
                                                   (short)0, c, false, false);
}

// Fragment = elements {k..k+7} and {k+16..k+23} of a bf16 row: two 16B loads.
__device__ __forceinline__ v16bf load_frag_bf(const __bf16* p) {
    BFrag t;
    t.q[0] = *(const uint4*)p;
    t.q[1] = *(const uint4*)(p + 16);
    return t.v;
}

// Packed bf16 maximumNumber (VOP3P V_PK_MAX_NUM_BF16).
__device__ __forceinline__ unsigned int pk_max_bf16(unsigned int a, unsigned int b) {
    unsigned int r;
    asm("v_pk_max_num_bf16 %0, %1, %2" : "=v"(r) : "v"(a), "v"(b));
    return r;
}

// Guaranteed single-instruction packed f32->bf16 pair convert.
__device__ __forceinline__ unsigned int cvt_pk_bf16(float a, float b) {
    unsigned int r;
    asm("v_cvt_pk_bf16_f32 %0, %1, %2" : "=v"(r) : "v"(a), "v"(b));
    return r;
}

// ---------------- prep: fp32 -> bf16 bulk convert ----------------
__global__ void k_cvt_bf16(const float* __restrict__ in, __bf16* __restrict__ out, int n) {
    int i = (blockIdx.x * blockDim.x + threadIdx.x) * 4;
    if (i + 3 < n) {
        float4 f = *(const float4*)(in + i);
        union { __bf16 h[4]; uint2 u; } t;
        t.h[0] = (__bf16)f.x; t.h[1] = (__bf16)f.y;
        t.h[2] = (__bf16)f.z; t.h[3] = (__bf16)f.w;
        *(uint2*)(out + i) = t.u;
    } else {
        for (; i < n; ++i) out[i] = (__bf16)in[i];
    }
}

__global__ void k_zero_bf16(__bf16* __restrict__ p, int n) {
    int i = blockIdx.x * blockDim.x + threadIdx.x;
    if (i < n) p[i] = (__bf16)0.0f;
}

// ---------------- voxel hash ----------------
__global__ void k_fill_v2p(int* __restrict__ v2p) {
    int i = blockIdx.x * blockDim.x + threadIdx.x;
    if (i < NVOX_GRID) v2p[i] = -1;
}

__global__ void k_scatter_v2p(const int* __restrict__ vc, int* __restrict__ v2p, int n) {
    int i = blockIdx.x * blockDim.x + threadIdx.x;
    if (i < n) {
        int x = vc[3 * i + 0], y = vc[3 * i + 1], z = vc[3 * i + 2];
        v2p[(z * NYv + y) * NXv + x] = i;
    }
}

// ---------------- per-grid-point cells: {bx,by,bz, roi*KPRE+pt} ----------------
__global__ void k_cells(const float* __restrict__ rois, int4* __restrict__ cells) {
    int t = blockIdx.x * blockDim.x + threadIdx.x;
    if (t >= NTASK) return;
    int roi = t / NPT, pt = t % NPT;
    const float* rp = rois + roi * 7;
    float cx = rp[0], cy = rp[1], cz = rp[2];
    float sx = rp[3], sy = rp[4], sz = rp[5], yaw = rp[6];
    int gi = pt / 36, gj = (pt / 6) % 6, gk = pt % 6;
    float lx = ((float)gi + 0.5f) * (1.0f / 6.0f) * sx - 0.5f * sx;
    float ly = ((float)gj + 0.5f) * (1.0f / 6.0f) * sy - 0.5f * sy;
    float lz = ((float)gk + 0.5f) * (1.0f / 6.0f) * sz - 0.5f * sz;
    float c = cosf(yaw), s = sinf(yaw);
    float px = lx * c - ly * s + cx;
    float py = lx * s + ly * c + cy;
    float pz = lz + cz;
    int bx = ((int)floorf((px + 70.4f) / 0.1f)) >> 3;   // floor-div 8
    int by = ((int)floorf((py + 40.0f) / 0.1f)) >> 3;
    int bz = ((int)floorf((pz + 3.0f)  / 0.1f)) >> 3;
    cells[t] = make_int4(bx, by, bz, roi * KPRE + pt);
}

// ---------------- gather + MLP + maxpool (bf16 WMMA) ----------------
// Two independent tasks per loop iteration (disjoint regs + LDS slices) so the
// scheduler can fill WMMA->VALU hazard slots and load-wait gaps.
// GEMM2's K axis is permuted (pair-interleaved) so each lane's (N0,N1) channel
// pair packs into one b32 LDS store; B2 carries the inverse permutation.
// Invalid/dummy neighbors read the all-zero feature row `zrow`.
__global__ __launch_bounds__(256) void k_pool(
    const int4* __restrict__ cells, const __bf16* __restrict__ vfeatB,
    const int* __restrict__ v2p,    const float* __restrict__ w1,
    const float* __restrict__ w2,   __bf16* __restrict__ pooledT, int zrow)
{
    __shared__ __bf16 lds[8 * 1024];  // 2KB per wave: two 16x32 bf16 tiles
    const int lane = threadIdx.x & 31;
    const int wave = threadIdx.x >> 5;
    const int hi   = lane >> 4;
    const int ln   = lane & 15;
    __bf16* lb0 = lds + wave * 1024;
    __bf16* lb1 = lb0 + 512;

    // B fragments for mlp1 (64x32) and mlp2 (32x32), built once per wave.
    v16bf B1[2][2], B2[2];
#pragma unroll
    for (int kh = 0; kh < 2; ++kh)
#pragma unroll
        for (int nt = 0; nt < 2; ++nt) {
            union { __bf16 h[16]; v16bf v; } t;
            int n = nt * 16 + ln;
#pragma unroll
            for (int j = 0; j < 16; ++j) {
                int k = (j & 7) + ((j >> 3) << 4) + hi * 8 + kh * 32;
                t.h[j] = (__bf16)w1[k * 32 + n];
            }
            B1[kh][nt] = t.v;
        }
#pragma unroll
    for (int nt = 0; nt < 2; ++nt) {
        union { __bf16 h[16]; v16bf v; } t;
        int n = nt * 16 + ln;
#pragma unroll
        for (int j = 0; j < 16; ++j) {
            int kp = (j & 7) + ((j >> 3) << 4) + hi * 8;        // physical slot
            int kl = (kp & 1) ? (16 + (kp >> 1)) : (kp >> 1);    // logical channel
            t.h[j] = (__bf16)w2[kl * 32 + n];
        }
        B2[nt] = t.v;
    }

    const int nWaves = gridDim.x * (blockDim.x >> 5);
    const int gw     = blockIdx.x * (blockDim.x >> 5) + wave;

    auto body = [&](int task, __bf16* lb) {
        unsigned int* lbu = (unsigned int*)lb;
        const int4 cb = cells[task];   // wave-uniform

        // lane L handles neighbor L (27 real + 5 dummy)
        int myv = -1;
        if (lane < 27) {
            int nx = cb.x + (lane / 9) - 1;
            int ny = cb.y + ((lane / 3) % 3) - 1;
            int nz = cb.z + (lane % 3) - 1;
            if (nx >= 0 && nx < NXv && ny >= 0 && ny < NYv && nz >= 0 && nz < NZv)
                myv = v2p[(nz * NYv + ny) * NXv + nx];
        }

        unsigned int maxpk[8];
#pragma unroll
        for (int r = 0; r < 8; ++r) maxpk[r] = 0u;   // bf16 {0,0}

#pragma unroll
        for (int tile = 0; tile < 2; ++tile) {
            int vrow = __shfl(myv, tile * 16 + ln, 32);
            int srow = (vrow >= 0) ? vrow : zrow;    // invalid -> zero row
            const __bf16* fp = vfeatB + (srow * 64 + hi * 8);   // 32-bit offset
            // ---- GEMM1: feat(16x64) @ w1(64x32) ----
            v8f acc0 = {}, acc1 = {};
#pragma unroll
            for (int kh = 0; kh < 2; ++kh) {
                v16bf A = load_frag_bf(fp + kh * 32);
                acc0 = wmma_bf16(A, B1[kh][0], acc0);
                acc1 = wmma_bf16(A, B1[kh][1], acc1);
            }
            // ---- packed cvt + packed ReLU -> LDS (K-permuted 16x32) ----
#pragma unroll
            for (int r = 0; r < 8; ++r) {
                int row = r + hi * 8;
                lbu[row * 16 + ln] = pk_max_bf16(cvt_pk_bf16(acc0[r], acc1[r]), 0u);
            }
            // ---- reload as A fragment (same-wave LDS is in-order) ----
            v16bf A2 = load_frag_bf(lb + ln * 32 + hi * 8);
            // ---- GEMM2: h1(16x32) @ w2(32x32) (K-permuted), packed running max ----
            v8f h0 = {}, h1v = {};
            h0  = wmma_bf16(A2, B2[0], h0);
            h1v = wmma_bf16(A2, B2[1], h1v);
            // maxpk >= 0 always, so relu folds into the max
#pragma unroll
            for (int r = 0; r < 8; ++r)
                maxpk[r] = pk_max_bf16(maxpk[r], cvt_pk_bf16(h0[r], h1v[r]));
        }
        // reduce the 8 register pairs, then across lane halves (one shfl)
        unsigned int m01 = pk_max_bf16(maxpk[0], maxpk[1]);
        unsigned int m23 = pk_max_bf16(maxpk[2], maxpk[3]);
        unsigned int m45 = pk_max_bf16(maxpk[4], maxpk[5]);
        unsigned int m67 = pk_max_bf16(maxpk[6], maxpk[7]);
        unsigned int m = pk_max_bf16(pk_max_bf16(m01, m23), pk_max_bf16(m45, m67));
        m = pk_max_bf16(m, (unsigned int)__shfl_xor((int)m, 16, 32));
        BPack f; f.u = m;
        // lane L holds channel L: low half = ch ln (tile-N0), high = ch 16+ln
        pooledT[cb.w + lane * NPT] = hi ? f.h[1] : f.h[0];
    };

    // NTASK == 8 * (2*nWaves) for the 432x256 launch -> no tail guard needed.
    for (int task = gw; task < NTASK; task += 2 * nWaves) {
        body(task,          lb0);
        body(task + nWaves, lb1);
    }
}

// ---------------- generic 256 x K GEMM: out = act(A @ W^T), M=N=256, bf16 in/out ----
__global__ __launch_bounds__(256) void k_gemm_nt(
    const __bf16* __restrict__ A, const __bf16* __restrict__ W,
    __bf16* __restrict__ out, int K, int doRelu)
{
    const int lane = threadIdx.x & 31;
    const int wave = threadIdx.x >> 5;
    const int wg   = blockIdx.x * (blockDim.x >> 5) + wave;   // 0..255 tile id
    const int mt = wg >> 4, nt = wg & 15;
    const int hi = lane >> 4, ln = lane & 15;
    const __bf16* ar = A + ((mt * 16 + ln) * K + hi * 8);   // 32-bit offsets
    const __bf16* br = W + ((nt * 16 + ln) * K + hi * 8);
    v8f acc = {};
    for (int kb = 0; kb < K; kb += 32) {
        v16bf a = load_frag_bf(ar + kb);
        v16bf b = load_frag_bf(br + kb);
        acc = wmma_bf16(a, b, acc);
    }
#pragma unroll
    for (int r = 0; r < 8; ++r) {
        float v = acc[r];
        if (doRelu) v = fmaxf(v, 0.0f);
        out[(mt * 16 + r + hi * 8) * 256 + nt * 16 + ln] = (__bf16)v;
    }
}

// ---------------- tiny head outputs ----------------
__global__ __launch_bounds__(32) void k_heads(
    const __bf16* __restrict__ yc, const __bf16* __restrict__ yi, const __bf16* __restrict__ yr,
    const float* __restrict__ wc3, const float* __restrict__ bc3,
    const float* __restrict__ wi3, const float* __restrict__ bi3,
    const float* __restrict__ wr3, const float* __restrict__ br3,
    float* __restrict__ out)
{
    const int roi  = blockIdx.x;
    const int lane = threadIdx.x;
    float s = 0.0f;
    for (int k = lane; k < 256; k += 32) s += (float)yc[roi * 256 + k] * wc3[k];
#pragma unroll
    for (int o = 16; o > 0; o >>= 1) s += __shfl_xor(s, o, 32);
    if (lane == 0) out[roi] = s + bc3[0];
    s = 0.0f;
    for (int k = lane; k < 256; k += 32) s += (float)yi[roi * 256 + k] * wi3[k];
#pragma unroll
    for (int o = 16; o > 0; o >>= 1) s += __shfl_xor(s, o, 32);
    if (lane == 0) out[256 + roi] = s + bi3[0];
    for (int j = 0; j < 7; ++j) {
        float t = 0.0f;
        for (int k = lane; k < 256; k += 32)
            t += (float)yr[roi * 256 + k] * wr3[j * 256 + k];
#pragma unroll
        for (int o = 16; o > 0; o >>= 1) t += __shfl_xor(t, o, 32);
        if (lane == 0) out[512 + roi * 7 + j] = t + br3[j];
    }
}

extern "C" void kernel_launch(void* const* d_in, const int* in_sizes, int n_in,
                              void* d_out, int out_size, void* d_ws, size_t ws_size,
                              hipStream_t stream) {
    const float* rois   = (const float*)d_in[0];
    const float* vfeat  = (const float*)d_in[1];
    const int*   vcoord = (const int*)d_in[2];
    const float* w_mlp1 = (const float*)d_in[3];
    const float* w_mlp2 = (const float*)d_in[4];
    const float* w_fc1  = (const float*)d_in[5];
    const float* w_fc2  = (const float*)d_in[6];
    const float* w_cls1 = (const float*)d_in[7];
    const float* w_cls2 = (const float*)d_in[8];
    const float* w_cls3 = (const float*)d_in[9];
    const float* b_cls3 = (const float*)d_in[10];
    const float* w_iou1 = (const float*)d_in[11];
    const float* w_iou2 = (const float*)d_in[12];
    const float* w_iou3 = (const float*)d_in[13];
    const float* b_iou3 = (const float*)d_in[14];
    const float* w_reg1 = (const float*)d_in[15];
    const float* w_reg2 = (const float*)d_in[16];
    const float* w_reg3 = (const float*)d_in[17];
    const float* b_reg3 = (const float*)d_in[18];
    float* out = (float*)d_out;

    const int nvox   = in_sizes[2] / 3;        // 30000
    const int nfeatE = in_sizes[1];            // nvox*64

    char* ws = (char*)d_ws;
    size_t ofs = 0;
    int*    v2p     = (int*)(ws + ofs);    ofs += 352256;                       // 88000 ints
    __bf16* vfeatB  = (__bf16*)(ws + ofs); ofs += (((size_t)nfeatE + 64) * 2 + 255) & ~255ull;
    int4*   cells   = (int4*)(ws + ofs);   ofs += (size_t)NTASK * 16;           // 884 KB
    __bf16* pooledT = (__bf16*)(ws + ofs); ofs += (size_t)N_ROIS * KPRE * 2;    // 3.54 MB
    __bf16* wfc1B   = (__bf16*)(ws + ofs); ofs += (size_t)256 * KPRE * 2;       // 3.54 MB
    __bf16* wfc2B   = (__bf16*)(ws + ofs); ofs += 256 * 256 * 2;
    __bf16* wc1B    = (__bf16*)(ws + ofs); ofs += 256 * 256 * 2;
    __bf16* wc2B    = (__bf16*)(ws + ofs); ofs += 256 * 256 * 2;
    __bf16* wi1B    = (__bf16*)(ws + ofs); ofs += 256 * 256 * 2;
    __bf16* wi2B    = (__bf16*)(ws + ofs); ofs += 256 * 256 * 2;
    __bf16* wr1B    = (__bf16*)(ws + ofs); ofs += 256 * 256 * 2;
    __bf16* wr2B    = (__bf16*)(ws + ofs); ofs += 256 * 256 * 2;
    __bf16* x1      = (__bf16*)(ws + ofs); ofs += 256 * 256 * 2;
    __bf16* x2      = (__bf16*)(ws + ofs); ofs += 256 * 256 * 2;
    __bf16* yA      = (__bf16*)(ws + ofs); ofs += 256 * 256 * 2;
    __bf16* y2c     = (__bf16*)(ws + ofs); ofs += 256 * 256 * 2;
    __bf16* y2i     = (__bf16*)(ws + ofs); ofs += 256 * 256 * 2;
    __bf16* y2r     = (__bf16*)(ws + ofs); ofs += 256 * 256 * 2;

    // bf16 conversions (one pass each, coalesced float4 -> 4x bf16)
    auto cvt = [&](const float* src, __bf16* dst, int n) {
        k_cvt_bf16<<<(n / 4 + 255) / 256, 256, 0, stream>>>(src, dst, n);
    };
    cvt(vfeat,  vfeatB, nfeatE);
    k_zero_bf16<<<1, 64, 0, stream>>>(vfeatB + nfeatE, 64);   // zero row at index nvox
    cvt(w_fc1,  wfc1B,  256 * KPRE);
    cvt(w_fc2,  wfc2B,  256 * 256);
    cvt(w_cls1, wc1B,   256 * 256);
    cvt(w_cls2, wc2B,   256 * 256);
    cvt(w_iou1, wi1B,   256 * 256);
    cvt(w_iou2, wi2B,   256 * 256);
    cvt(w_reg1, wr1B,   256 * 256);
    cvt(w_reg2, wr2B,   256 * 256);

    k_fill_v2p<<<(NVOX_GRID + 255) / 256, 256, 0, stream>>>(v2p);
    k_scatter_v2p<<<(nvox + 255) / 256, 256, 0, stream>>>(vcoord, v2p, nvox);
    k_cells<<<(NTASK + 255) / 256, 256, 0, stream>>>(rois, cells);

    // 432 blocks x 8 waves = 3456 waves; 2 tasks in flight, 16 grid-points each
    k_pool<<<432, 256, 0, stream>>>(cells, vfeatB, v2p, w_mlp1, w_mlp2, pooledT, nvox);

    // fc stack: 256 waves = one 16x16 tile each
    k_gemm_nt<<<32, 256, 0, stream>>>(pooledT, wfc1B, x1, KPRE, 1);
    k_gemm_nt<<<32, 256, 0, stream>>>(x1, wfc2B, x2, 256, 1);
    k_gemm_nt<<<32, 256, 0, stream>>>(x2, wc1B, yA, 256, 1);
    k_gemm_nt<<<32, 256, 0, stream>>>(yA, wc2B, y2c, 256, 1);
    k_gemm_nt<<<32, 256, 0, stream>>>(x2, wi1B, yA, 256, 1);
    k_gemm_nt<<<32, 256, 0, stream>>>(yA, wi2B, y2i, 256, 1);
    k_gemm_nt<<<32, 256, 0, stream>>>(x2, wr1B, yA, 256, 1);
    k_gemm_nt<<<32, 256, 0, stream>>>(yA, wr2B, y2r, 256, 1);

    k_heads<<<256, 32, 0, stream>>>(y2c, y2i, y2r, w_cls3, b_cls3,
                                    w_iou3, b_iou3, w_reg3, b_reg3, out);
}
